// TransformerBlock_12945031430763
// MI455X (gfx1250) — compile-verified, matrix operand order checked by reading, not run on hardware
//
#include <hip/hip_runtime.h>
#include <cstdint>
#include <cstddef>

typedef __attribute__((ext_vector_type(16))) __bf16 v16bf;
typedef __attribute__((ext_vector_type(8)))  float  v8f;
typedef __attribute__((ext_vector_type(4)))  int    v4i;

#define DIMSZ   1024
#define HEADS   16
#define HEAD_D  64
#define HIDDEN  4096
#define BATCH   8
#define SEQ     1024
#define ROWS    (BATCH * SEQ)   /* 8192 */

#define AS1 __attribute__((address_space(1)))
#define AS3 __attribute__((address_space(3)))

#if __has_builtin(__builtin_amdgcn_global_load_async_to_lds_b128) && \
    __has_builtin(__builtin_amdgcn_s_wait_asynccnt)
#define USE_ASYNC_LDS 1
#else
#define USE_ASYNC_LDS 0
#endif

union FragBF {             // 32-byte WMMA 16-bit operand (two b128 loads)
  uint4 u[2];
  v16bf v;
};
union U8BF {               // 16-byte chunk viewed as 8 bf16
  uint4  u;
  __bf16 h[8];
};

// ------------------------------------------------------- async copy helpers
__device__ __forceinline__ void copy16_to_lds(const __bf16* g, __bf16* l) {
#if USE_ASYNC_LDS
  __builtin_amdgcn_global_load_async_to_lds_b128(
      (AS1 v4i*)g, (AS3 v4i*)l, 0, 0);
#else
  *(uint4*)l = *(const uint4*)g;
#endif
}
__device__ __forceinline__ void wait_async_lds() {
#if USE_ASYNC_LDS
  __builtin_amdgcn_s_wait_asynccnt(0);
#endif
}

// ------------------------------------------------------- DPP 16-lane reduce
// max / sum across each 16-lane row group (pure VALU, no LDS traffic)
#if __has_builtin(__builtin_amdgcn_update_dpp)
template <int CTRL>
__device__ __forceinline__ float dpp_bcast(float x) {
  int p = __builtin_amdgcn_update_dpp(__float_as_int(x), __float_as_int(x),
                                      CTRL, 0xF, 0xF, false);
  return __int_as_float(p);
}
__device__ __forceinline__ float red16_max(float x) {
  x = fmaxf(x, dpp_bcast<0xB1>(x));    // quad_perm [1,0,3,2]
  x = fmaxf(x, dpp_bcast<0x4E>(x));    // quad_perm [2,3,0,1]
  x = fmaxf(x, dpp_bcast<0x141>(x));   // row_half_mirror
  x = fmaxf(x, dpp_bcast<0x140>(x));   // row_mirror
  return x;
}
__device__ __forceinline__ float red16_sum(float x) {
  x += dpp_bcast<0xB1>(x);
  x += dpp_bcast<0x4E>(x);
  x += dpp_bcast<0x141>(x);
  x += dpp_bcast<0x140>(x);
  return x;
}
#else
__device__ __forceinline__ float red16_max(float x) {
  x = fmaxf(x, __shfl_xor(x, 1));
  x = fmaxf(x, __shfl_xor(x, 2));
  x = fmaxf(x, __shfl_xor(x, 4));
  x = fmaxf(x, __shfl_xor(x, 8));
  return x;
}
__device__ __forceinline__ float red16_sum(float x) {
  x += __shfl_xor(x, 1);
  x += __shfl_xor(x, 2);
  x += __shfl_xor(x, 4);
  x += __shfl_xor(x, 8);
  return x;
}
#endif

// ------------------------------------------------- f32 -> bf16 + transpose
// src[K][N] (f32, row-major) -> dst[N][K] (bf16). One-time weight prep so
// every GEMM can stage B with contiguous (async) 16B chunks.
__global__ __launch_bounds__(256)
void cvt_transpose_bf16_kernel(const float* __restrict__ src,
                               __bf16* __restrict__ dst, int K, int N) {
  __shared__ __bf16 tile[32][33];
  const int bn = blockIdx.x * 32;
  const int bk = blockIdx.y * 32;
  const int tx = threadIdx.x & 31, ty = threadIdx.x >> 5;  // 32 x 8
#pragma unroll
  for (int i = 0; i < 32; i += 8)
    tile[ty + i][tx] = (__bf16)src[(size_t)(bk + ty + i) * N + bn + tx];
  __syncthreads();
#pragma unroll
  for (int i = 0; i < 32; i += 8)
    dst[(size_t)(bn + ty + i) * K + bk + tx] = tile[tx][ty + i];
}

// ---------------------------------------------------------------- LayerNorm
__global__ __launch_bounds__(256)
void layernorm_bf16_kernel(const float* __restrict__ x,
                           const float* __restrict__ g,
                           const float* __restrict__ b,
                           __bf16* __restrict__ out) {
  __shared__ float s1[256], s2[256];
  const int row = blockIdx.x;
  const float* xr = x + (size_t)row * DIMSZ;
  float vals[4];
  float sum = 0.f, sq = 0.f;
#pragma unroll
  for (int i = 0; i < 4; ++i) {
    float v = xr[threadIdx.x + i * 256];
    vals[i] = v; sum += v; sq += v * v;
  }
  s1[threadIdx.x] = sum; s2[threadIdx.x] = sq;
  __syncthreads();
  for (int off = 128; off > 0; off >>= 1) {
    if (threadIdx.x < off) {
      s1[threadIdx.x] += s1[threadIdx.x + off];
      s2[threadIdx.x] += s2[threadIdx.x + off];
    }
    __syncthreads();
  }
  const float mu  = s1[0] * (1.f / DIMSZ);
  const float var = s2[0] * (1.f / DIMSZ) - mu * mu;
  const float rs  = rsqrtf(var + 1e-6f);
#pragma unroll
  for (int i = 0; i < 4; ++i) {
    int c = threadIdx.x + i * 256;
    out[(size_t)row * DIMSZ + c] = (__bf16)((vals[i] - mu) * rs * g[c] + b[c]);
  }
}

// ---------------------------------------------------------------- WMMA GEMM
// C[M,N] = A[M,K](bf16) x Bt[N,K](bf16, pre-transposed weights) + bias.
// EPI: 0 = bias -> bf16 | 1 = bias + residual(f32) -> f32 | 2 = bias+GELU -> bf16
// 256 threads = 8 waves (4 M x 2 N); WG tile 128x128, K-step 32,
// double-buffered LDS, async global->LDS staging, 1 barrier / K-step.
template <int EPI>
__global__ __launch_bounds__(256)
void gemm_wmma_kernel(const __bf16* __restrict__ A, const __bf16* __restrict__ Bt,
                      const float* __restrict__ bias, const float* __restrict__ res,
                      float* __restrict__ outF, __bf16* __restrict__ outB,
                      int M, int N, int K) {
  __shared__ __bf16 lds_a[2][128 * 40];   // [row][k], pitch 40 bf16 (conflict-free)
  __shared__ __bf16 lds_b[2][128 * 40];   // [n][k]
  const int tid  = threadIdx.x;
  const int lane = tid & 31;
  const int wave = tid >> 5;
  const int wm   = wave & 3;
  const int wn   = wave >> 2;
  const int row0 = blockIdx.y * 128;
  const int col0 = blockIdx.x * 128;
  const int lr   = lane & 15;
  const int kb   = (lane >> 4) * 8;
  const int rowHalf = (lane >> 4) * 8;

  const v8f vzero = {0.f, 0.f, 0.f, 0.f, 0.f, 0.f, 0.f, 0.f};
  v8f acc[2][4];
#pragma unroll
  for (int mi = 0; mi < 2; ++mi)
#pragma unroll
    for (int ni = 0; ni < 4; ++ni) acc[mi][ni] = vzero;

  // stage one 128x32 tile of A and Bt into buffer `buf`
  auto stage = [&](int buf, int k0) {
    for (int c = tid; c < 512; c += 256) {       // A: 128 rows x 4 chunks
      int r = c >> 2, cc = (c & 3) * 8;
      copy16_to_lds(&A[(size_t)(row0 + r) * K + k0 + cc],
                    &lds_a[buf][r * 40 + cc]);
    }
    for (int c = tid; c < 512; c += 256) {       // Bt: 128 cols x 4 chunks
      int r = c >> 2, cc = (c & 3) * 8;
      copy16_to_lds(&Bt[(size_t)(col0 + r) * K + k0 + cc],
                    &lds_b[buf][r * 40 + cc]);
    }
  };

  const int nk = K >> 5;
  stage(0, 0);
  for (int kt = 0; kt < nk; ++kt) {
    const int buf = kt & 1;
    wait_async_lds();
    __syncthreads();
    if (kt + 1 < nk) stage(buf ^ 1, (kt + 1) << 5);

    FragBF af[2], bfr[4];
#pragma unroll
    for (int mi = 0; mi < 2; ++mi) {
      int r = wm * 32 + mi * 16 + lr;
      af[mi].u[0] = *(const uint4*)&lds_a[buf][r * 40 + kb];
      af[mi].u[1] = *(const uint4*)&lds_a[buf][r * 40 + kb + 16];
    }
#pragma unroll
    for (int ni = 0; ni < 4; ++ni) {
      int n = wn * 64 + ni * 16 + lr;
      bfr[ni].u[0] = *(const uint4*)&lds_b[buf][n * 40 + kb];
      bfr[ni].u[1] = *(const uint4*)&lds_b[buf][n * 40 + kb + 16];
    }
#pragma unroll
    for (int mi = 0; mi < 2; ++mi)
#pragma unroll
      for (int ni = 0; ni < 4; ++ni)
        acc[mi][ni] = __builtin_amdgcn_wmma_f32_16x16x32_bf16(
            false, af[mi].v, false, bfr[ni].v, (short)0, acc[mi][ni], false, false);
  }

  // fused epilogue
#pragma unroll
  for (int mi = 0; mi < 2; ++mi) {
#pragma unroll
    for (int ni = 0; ni < 4; ++ni) {
      const int col = col0 + wn * 64 + ni * 16 + lr;
      const float bv = bias[col];
#pragma unroll
      for (int v = 0; v < 8; ++v) {
        const int row = row0 + wm * 32 + mi * 16 + v + rowHalf;
        float val = acc[mi][ni][v] + bv;
        const size_t idx = (size_t)row * N + col;
        if (EPI == 1) {
          outF[idx] = val + res[idx];
        } else if (EPI == 2) {
          val = 0.5f * val * (1.f + erff(val * 0.70710678118654752f));
          outB[idx] = (__bf16)val;
        } else {
          outB[idx] = (__bf16)val;
        }
      }
    }
  }
}

// ---------------------------------------------------------------- attention
// qkv: bf16 [B, N, 3, HEADS, HEAD_D]; out: bf16 [B, N, DIM]
// 128 threads = 4 waves; each wave owns 16 query rows; block covers 64 rows
// of one (b,h). Flash-style online softmax over 32-key chunks.
__global__ __launch_bounds__(128)
void attention_wmma_kernel(const __bf16* __restrict__ qkv,
                           __bf16* __restrict__ out) {
  __shared__ __bf16 lds_k[32 * 72];        // [kv][hd], pitch 72
  __shared__ __bf16 lds_v[64 * 40];        // [hd][kv], pitch 40 (transposed)
  __shared__ __bf16 lds_p[4 * 16 * 40];    // per-wave P C->A relayout patch

  const int bh = blockIdx.x >> 4;
  const int qt = blockIdx.x & 15;
  const int b  = bh >> 4;
  const int h  = bh & 15;
  const int tid  = threadIdx.x;
  const int lane = tid & 31;
  const int wave = tid >> 5;
  const int lr   = lane & 15;
  const int kb   = (lane >> 4) * 8;
  const int rowHalf = (lane >> 4) * 8;
  const int qbase = qt * 64 + wave * 16;

  const size_t qrow = (size_t)(b * SEQ + qbase + lr) * (3 * DIMSZ) + h * HEAD_D;
  FragBF qf[2];
#pragma unroll
  for (int f = 0; f < 2; ++f) {
    qf[f].u[0] = *(const uint4*)&qkv[qrow + f * 32 + kb];
    qf[f].u[1] = *(const uint4*)&qkv[qrow + f * 32 + kb + 16];
  }

  const v8f vzero = {0.f, 0.f, 0.f, 0.f, 0.f, 0.f, 0.f, 0.f};
  v8f o[4];
#pragma unroll
  for (int ni = 0; ni < 4; ++ni) o[ni] = vzero;
  float m[8], l[8];
#pragma unroll
  for (int v = 0; v < 8; ++v) { m[v] = -1e30f; l[v] = 0.f; }

  for (int kv0 = 0; kv0 < SEQ; kv0 += 32) {
    // K chunk [32][64] -> lds_k[kv][hd] (async, no relayout needed)
    for (int c = tid; c < 256; c += 128) {
      int r = c >> 3, cc = (c & 7) * 8;
      const size_t src = (size_t)(b * SEQ + kv0 + r) * (3 * DIMSZ)
                         + DIMSZ + h * HEAD_D + cc;
      copy16_to_lds(&qkv[src], &lds_k[r * 72 + cc]);
    }
    // V chunk transposed -> lds_v[hd][kv]
    for (int c = tid; c < 256; c += 128) {
      int r = c >> 3, cc = (c & 7) * 8;
      const size_t src = (size_t)(b * SEQ + kv0 + r) * (3 * DIMSZ)
                         + 2 * DIMSZ + h * HEAD_D + cc;
      U8BF t; t.u = *(const uint4*)&qkv[src];
#pragma unroll
      for (int j = 0; j < 8; ++j) lds_v[(cc + j) * 40 + r] = t.h[j];
    }
    wait_async_lds();
    __syncthreads();

    // scores: two 16x16 tiles (kv 0..15, 16..31), K-dim = hd (2 x 32)
    v8f s0 = vzero, s1 = vzero;
#pragma unroll
    for (int f = 0; f < 2; ++f) {
      FragBF kf0, kf1;
      const int base0 = lr * 72 + f * 32 + kb;
      const int base1 = (16 + lr) * 72 + f * 32 + kb;
      kf0.u[0] = *(const uint4*)&lds_k[base0];
      kf0.u[1] = *(const uint4*)&lds_k[base0 + 16];
      kf1.u[0] = *(const uint4*)&lds_k[base1];
      kf1.u[1] = *(const uint4*)&lds_k[base1 + 16];
      s0 = __builtin_amdgcn_wmma_f32_16x16x32_bf16(false, qf[f].v, false, kf0.v,
                                                   (short)0, s0, false, false);
      s1 = __builtin_amdgcn_wmma_f32_16x16x32_bf16(false, qf[f].v, false, kf1.v,
                                                   (short)0, s1, false, false);
    }

    // online softmax (DPP row reductions); write P bf16 to per-wave patch
    const float scale = 0.125f;   // 1/sqrt(64)
    __bf16* pw = &lds_p[wave * 16 * 40];
#pragma unroll
    for (int v = 0; v < 8; ++v) {
      const float a0 = s0[v] * scale, a1 = s1[v] * scale;
      const float mx = red16_max(fmaxf(a0, a1));
      const float mn    = fmaxf(m[v], mx);
      const float alpha = __expf(m[v] - mn);
      const float p0 = __expf(a0 - mn);
      const float p1 = __expf(a1 - mn);
      const float sum = red16_sum(p0 + p1);
      l[v] = l[v] * alpha + sum;
      m[v] = mn;
#pragma unroll
      for (int ni = 0; ni < 4; ++ni) o[ni][v] *= alpha;
      const int prow = v + rowHalf;
      pw[prow * 40 + lr]      = (__bf16)p0;
      pw[prow * 40 + 16 + lr] = (__bf16)p1;
    }

    // P (A operand) x V (B operand): K-dim = 32 keys
    FragBF pf;
    pf.u[0] = *(const uint4*)&pw[lr * 40 + kb];
    pf.u[1] = *(const uint4*)&pw[lr * 40 + kb + 16];
#pragma unroll
    for (int ni = 0; ni < 4; ++ni) {
      FragBF vf;
      const int n = ni * 16 + lr;
      vf.u[0] = *(const uint4*)&lds_v[n * 40 + kb];
      vf.u[1] = *(const uint4*)&lds_v[n * 40 + kb + 16];
      o[ni] = __builtin_amdgcn_wmma_f32_16x16x32_bf16(false, pf.v, false, vf.v,
                                                      (short)0, o[ni], false, false);
    }
    __syncthreads();
  }

  // normalize and store bf16 [b, n, h*64 + d]
#pragma unroll
  for (int v = 0; v < 8; ++v) {
    const float inv = 1.f / l[v];
    const int row = qbase + v + rowHalf;
#pragma unroll
    for (int ni = 0; ni < 4; ++ni) {
      const int col = h * HEAD_D + ni * 16 + lr;
      out[(size_t)(b * SEQ + row) * DIMSZ + col] = (__bf16)(o[ni][v] * inv);
    }
  }
}

// ---------------------------------------------------------------- launch
extern "C" void kernel_launch(void* const* d_in, const int* in_sizes, int n_in,
                              void* d_out, int out_size, void* d_ws, size_t ws_size,
                              hipStream_t stream) {
  (void)in_sizes; (void)n_in; (void)out_size; (void)ws_size;
  const float* x      = (const float*)d_in[0];
  const float* ln_g   = (const float*)d_in[1];
  const float* ln_b   = (const float*)d_in[2];
  const float* w_qkv  = (const float*)d_in[3];
  const float* b_qkv  = (const float*)d_in[4];
  const float* w_proj = (const float*)d_in[5];
  const float* b_proj = (const float*)d_in[6];
  const float* w_fc1  = (const float*)d_in[7];
  const float* b_fc1  = (const float*)d_in[8];
  const float* w_fc2  = (const float*)d_in[9];
  const float* b_fc2  = (const float*)d_in[10];
  float* out = (float*)d_out;

  // workspace layout (bytes); weights stored TRANSPOSED [N][K] in bf16
  char* ws = (char*)d_ws;
  __bf16* wqkv_t = (__bf16*)(ws + 0);                    //  6 291 456 B
  __bf16* wprj_t = (__bf16*)(ws + 6291456);              //  2 097 152 B
  __bf16* wfc1_t = (__bf16*)(ws + 8388608);              //  8 388 608 B
  __bf16* wfc2_t = (__bf16*)(ws + 16777216);             //  8 388 608 B
  __bf16* xn_b   = (__bf16*)(ws + 25165824);             // 16 777 216 B (xn, later y)
  __bf16* qkv_b  = (__bf16*)(ws + 41943040);             // 50 331 648 B
  __bf16* attn_b = (__bf16*)(ws + 92274688);             // 16 777 216 B
  float*  h_f    = (float*) (ws + 109051904);            // 33 554 432 B
  __bf16* act1_b = (__bf16*)(ws + 41943040);             // reuses qkv+attn (64 MiB)

  // 1) weights f32 -> bf16, transposed once (B-tiles become contiguous)
  cvt_transpose_bf16_kernel<<<dim3(3 * DIMSZ / 32, DIMSZ / 32), 256, 0, stream>>>(
      w_qkv, wqkv_t, DIMSZ, 3 * DIMSZ);
  cvt_transpose_bf16_kernel<<<dim3(DIMSZ / 32, DIMSZ / 32), 256, 0, stream>>>(
      w_proj, wprj_t, DIMSZ, DIMSZ);
  cvt_transpose_bf16_kernel<<<dim3(HIDDEN / 32, DIMSZ / 32), 256, 0, stream>>>(
      w_fc1, wfc1_t, DIMSZ, HIDDEN);
  cvt_transpose_bf16_kernel<<<dim3(DIMSZ / 32, HIDDEN / 32), 256, 0, stream>>>(
      w_fc2, wfc2_t, HIDDEN, DIMSZ);

  // 2) xn = LN(x)
  layernorm_bf16_kernel<<<ROWS, 256, 0, stream>>>(x, ln_g, ln_b, xn_b);

  // 3) qkv = xn @ w_qkv + b_qkv  (bf16 out)
  gemm_wmma_kernel<0><<<dim3(3 * DIMSZ / 128, ROWS / 128), 256, 0, stream>>>(
      xn_b, wqkv_t, b_qkv, nullptr, nullptr, qkv_b, ROWS, 3 * DIMSZ, DIMSZ);

  // 4) attention
  attention_wmma_kernel<<<BATCH * HEADS * (SEQ / 64), 128, 0, stream>>>(qkv_b, attn_b);

  // 5) h = x + attn @ w_proj + b_proj  (f32 out)
  gemm_wmma_kernel<1><<<dim3(DIMSZ / 128, ROWS / 128), 256, 0, stream>>>(
      attn_b, wprj_t, b_proj, x, h_f, nullptr, ROWS, DIMSZ, DIMSZ);

  // 6) y = LN(h)
  layernorm_bf16_kernel<<<ROWS, 256, 0, stream>>>(h_f, ln_g, ln_b, xn_b);

  // 7) act1 = gelu(y @ w_fc1 + b_fc1)  (bf16 out)
  gemm_wmma_kernel<2><<<dim3(HIDDEN / 128, ROWS / 128), 256, 0, stream>>>(
      xn_b, wfc1_t, b_fc1, nullptr, nullptr, act1_b, ROWS, HIDDEN, DIMSZ);

  // 8) out = h + act1 @ w_fc2 + b_fc2  (f32 out)
  gemm_wmma_kernel<1><<<dim3(DIMSZ / 128, ROWS / 128), 256, 0, stream>>>(
      act1_b, wfc2_t, b_fc2, h_f, out, nullptr, ROWS, DIMSZ, HIDDEN);
}
